// POISEVAE_50259707297995
// MI455X (gfx1250) — matrix operand order repaired and checked by reading, not compiled, not run on hardware
//
#include <hip/hip_runtime.h>

typedef __attribute__((ext_vector_type(16))) __bf16 v16bf;
typedef __attribute__((ext_vector_type(8)))  float  v8f;

// ---------------------------------------------------------------- RNG helpers
__device__ __forceinline__ unsigned hmix(unsigned x){
  x ^= x >> 16; x *= 0x7feb352dU;
  x ^= x >> 15; x *= 0x846ca68bU;
  x ^= x >> 16; return x;
}
// one draw (used for init only)
__device__ __forceinline__ float gaussf(unsigned s){
  unsigned a = hmix(s ^ 0x9E3779B9U);
  unsigned b = hmix(a + 0x85EBCA6BU);
  float u1 = (float)((a >> 8) | 1u) * 5.9604645e-08f;
  float u2 = (float)(b >> 8) * 5.9604645e-08f;
  float r  = __builtin_amdgcn_sqrtf(fmaxf(-2.0f * __logf(u1), 0.0f));
  return r * __cosf(6.28318530718f * u2);
}
// two draws per Box-Muller (halves transcendental count in the hot loop)
__device__ __forceinline__ float2 gauss2(unsigned s){
  unsigned a = hmix(s ^ 0x9E3779B9U);
  unsigned b = hmix(a + 0x85EBCA6BU);
  float u1 = (float)((a >> 8) | 1u) * 5.9604645e-08f;
  float u2 = (float)(b >> 8) * 5.9604645e-08f;
  float r  = __builtin_amdgcn_sqrtf(fmaxf(-2.0f * __logf(u1), 0.0f));
  float sn, cs;
  __sincosf(6.28318530718f * u2, &sn, &cs);
  return make_float2(r * cs, r * sn);
}

// ---------------------------------------------------------------- GEMM kernel
// C[M,N] = act(A[M,K] @ B[K,N] + bias[N]) ; f32 in memory, bf16 WMMA math.
// 256 threads (8 waves), 64x64 WG tile, K-panel = 64. B staged transposed so
// all fragment loads are contiguous ds_load_b128 pairs.
enum { ACT_NONE = 0, ACT_RELU = 1, ACT_NEGEXP = 2 };

__global__ __launch_bounds__(256)
void gemm_bias_act(const float* __restrict__ A, const float* __restrict__ B,
                   const float* __restrict__ bias, float* __restrict__ C,
                   int M, int K, int N, int act)
{
  __shared__ __bf16 sA [64 * 64];   // [m][k]
  __shared__ __bf16 sBt[64 * 64];   // [n][k]  (transposed panel)
  const int tid  = threadIdx.x;
  const int lane = tid & 31, w = tid >> 5;
  const int m0 = blockIdx.y * 64, n0 = blockIdx.x * 64;
  const int nt  = (w & 3) * 16;
  const int mtB = (w >> 2) * 32;

  const int mA  = lane & 15;
  const int kbA = (lane & 16) ? 8 : 0;
  const int nB  = nt + (lane & 15);
  const int kbB = (lane & 16) ? 8 : 0;

  v8f c0 = {}, c1 = {};
  for (int k0 = 0; k0 < K; k0 += 64){
    __syncthreads();
    { // stage A panel 64x64 (f32 -> bf16), row-major; 16 elems/thread
      int idx = tid * 16, r = idx >> 6, cc = idx & 63;
      const float* g = A + (m0 + r) * (long)K + k0 + cc;
      if (k0 + 64 < K) __builtin_prefetch(g + 64, 0, 1);
      #pragma unroll
      for (int j = 0; j < 16; ++j) sA[r * 64 + cc + j] = (__bf16)g[j];
    }
    { // stage B panel 64x64 transposed -> sBt[n][k]
      int idx = tid * 16, r = idx >> 6, cc = idx & 63;
      const float* g = B + (k0 + r) * (long)N + n0 + cc;
      if (k0 + 64 < K) __builtin_prefetch(g + 64 * (long)N, 0, 1);
      #pragma unroll
      for (int j = 0; j < 16; ++j) sBt[(cc + j) * 64 + r] = (__bf16)g[j];
    }
    __syncthreads();

    #pragma unroll
    for (int ks = 0; ks < 64; ks += 32){
      v16bf a0, a1, bfr;
      const __bf16* p0 = sA  + (mtB + mA) * 64 + ks + kbA;
      const __bf16* p1 = sA  + (mtB + 16 + mA) * 64 + ks + kbA;
      const __bf16* pb = sBt + nB * 64 + ks + kbB;
      #pragma unroll
      for (int j = 0; j < 8; ++j){
        a0[j] = p0[j];  a0[8 + j] = p0[16 + j];
        a1[j] = p1[j];  a1[8 + j] = p1[16 + j];
      }
      #pragma unroll
      for (int j = 0; j < 8; ++j){ bfr[j] = pb[j]; bfr[8 + j] = pb[16 + j]; }

      c0 = __builtin_amdgcn_wmma_f32_16x16x32_bf16(false, a0, false, bfr, (short)0, c0, false, false);
      c1 = __builtin_amdgcn_wmma_f32_16x16x32_bf16(false, a1, false, bfr, (short)0, c1, false, false);
    }
  }

  const int nn  = n0 + nB;
  const int mof = (lane & 16) ? 8 : 0;
  const float bv = bias ? bias[nn] : 0.0f;
  #pragma unroll
  for (int i = 0; i < 8; ++i){
    float v0 = c0[i] + bv;
    float v1 = c1[i] + bv;
    if (act == ACT_RELU)   { v0 = fmaxf(v0, 0.0f); v1 = fmaxf(v1, 0.0f); }
    if (act == ACT_NEGEXP) { v0 = -__expf(v0);     v1 = -__expf(v1); }
    C[(m0 + mtB + mof + i)      * (long)N + nn] = v0;
    C[(m0 + mtB + 16 + mof + i) * (long)N + nn] = v1;
  }
}

// ---------------------------------------------------------------- Gibbs chain
// One half-step: dstZ = sample( srcZ @ G1' + lmu , srcZ^2 @ G2' + lvar ).
// G1/G2: bf16 LDS, B(k,n) = G[n*128+k] contiguous (z1|z2: row-major copies,
// z2|z1: transposed copies). z and z^2 live in LDS as bf16; f32 z carried in
// registers. lam values (lmu/lvar, 16 each) and the Gaussian draws nr[8] are
// precomputed by the caller so only rcp/sqrt/fma remain after the WMMAs.
__device__ __forceinline__ void gibbs_half(
    const __bf16* __restrict__ srcZb, const __bf16* __restrict__ srcZq,
    __bf16* __restrict__ dstZb, __bf16* __restrict__ dstZq, float* zf,
    const __bf16* __restrict__ G1, const __bf16* __restrict__ G2,
    const float* lmu, const float* lvar,     // 16 register-resident values
    const float2* nr,                        // 8 precomputed normal pairs
    int lane, int w)
{
  const int nOff = w * 16;
  const int mA  = lane & 15;
  const int kbA = (lane & 16) ? 8 : 0;
  const int nB  = nOff + (lane & 15);
  const int kbB = (lane & 16) ? 8 : 0;

  v8f e1a = {}, e1b = {}, e2a = {}, e2b = {};
  #pragma unroll
  for (int k0 = 0; k0 < 128; k0 += 32){
    v16bf a0, a1, q0, q1, bg, bq;
    const __bf16* pa0 = srcZb + mA * 128 + k0 + kbA;
    const __bf16* pa1 = srcZb + (16 + mA) * 128 + k0 + kbA;
    const __bf16* pq0 = srcZq + mA * 128 + k0 + kbA;
    const __bf16* pq1 = srcZq + (16 + mA) * 128 + k0 + kbA;
    #pragma unroll
    for (int j = 0; j < 8; ++j){
      a0[j] = pa0[j]; a0[8 + j] = pa0[16 + j];
      a1[j] = pa1[j]; a1[8 + j] = pa1[16 + j];
      q0[j] = pq0[j]; q0[8 + j] = pq0[16 + j];
      q1[j] = pq1[j]; q1[8 + j] = pq1[16 + j];
    }
    const __bf16* pb1 = G1 + nB * 128 + k0 + kbB;
    const __bf16* pb2 = G2 + nB * 128 + k0 + kbB;
    #pragma unroll
    for (int j = 0; j < 8; ++j){ bg[j] = pb1[j]; bg[8 + j] = pb1[16 + j]; bq[j] = pb2[j]; bq[8 + j] = pb2[16 + j]; }

    e1a = __builtin_amdgcn_wmma_f32_16x16x32_bf16(false, a0, false, bg, (short)0, e1a, false, false);
    e1b = __builtin_amdgcn_wmma_f32_16x16x32_bf16(false, a1, false, bg, (short)0, e1b, false, false);
    e2a = __builtin_amdgcn_wmma_f32_16x16x32_bf16(false, q0, false, bq, (short)0, e2a, false, false);
    e2b = __builtin_amdgcn_wmma_f32_16x16x32_bf16(false, q1, false, bq, (short)0, e2b, false, false);
  }

  const int colL = nOff + (lane & 15);
  const int mof  = (lane & 16) ? 8 : 0;
  #pragma unroll
  for (int i = 0; i < 8; i += 2){
    { // row tile 0: two consecutive rows share one Box-Muller pair
      int M = mof + i;
      float2 g = nr[i >> 1];
      float eta1 = e1a[i]     + lmu[i];
      float eta2 = e2a[i]     + lvar[i];
      float vv = -0.5f * __builtin_amdgcn_rcpf(eta2);            // eta2 < 0
      float zn = eta1 * vv + __builtin_amdgcn_sqrtf(fmaxf(vv, 0.0f)) * g.x;
      zf[i] = zn;
      dstZb[M * 128 + colL] = (__bf16)zn;
      dstZq[M * 128 + colL] = (__bf16)(zn * zn);

      float eta1b = e1a[i + 1] + lmu[i + 1];
      float eta2b = e2a[i + 1] + lvar[i + 1];
      float vvb = -0.5f * __builtin_amdgcn_rcpf(eta2b);
      float znb = eta1b * vvb + __builtin_amdgcn_sqrtf(fmaxf(vvb, 0.0f)) * g.y;
      zf[i + 1] = znb;
      dstZb[(M + 1) * 128 + colL] = (__bf16)znb;
      dstZq[(M + 1) * 128 + colL] = (__bf16)(znb * znb);
    }
    { // row tile 1
      int M = 16 + mof + i;
      float2 g = nr[4 + (i >> 1)];
      float eta1 = e1b[i]     + lmu[8 + i];
      float eta2 = e2b[i]     + lvar[8 + i];
      float vv = -0.5f * __builtin_amdgcn_rcpf(eta2);
      float zn = eta1 * vv + __builtin_amdgcn_sqrtf(fmaxf(vv, 0.0f)) * g.x;
      zf[8 + i] = zn;
      dstZb[M * 128 + colL] = (__bf16)zn;
      dstZq[M * 128 + colL] = (__bf16)(zn * zn);

      float eta1b = e1b[i + 1] + lmu[8 + i + 1];
      float eta2b = e2b[i + 1] + lvar[8 + i + 1];
      float vvb = -0.5f * __builtin_amdgcn_rcpf(eta2b);
      float znb = eta1b * vvb + __builtin_amdgcn_sqrtf(fmaxf(vvb, 0.0f)) * g.y;
      zf[8 + i + 1] = znb;
      dstZb[(M + 1) * 128 + colL] = (__bf16)znb;
      dstZq[(M + 1) * 128 + colL] = (__bf16)(znb * znb);
    }
  }
}

__global__ __launch_bounds__(256)
void gibbs_kernel(const float* __restrict__ g11, const float* __restrict__ g22,
                  const float* __restrict__ mu1, const float* __restrict__ var1,
                  const float* __restrict__ mu2, const float* __restrict__ var2,
                  float* __restrict__ zp1, float* __restrict__ zp2,
                  float* __restrict__ zq1, float* __restrict__ zq2,
                  const int* __restrict__ nInitP)
{
  __shared__ __bf16 sG11 [128 * 128];  // g11 row-major       (z1|z2 step)
  __shared__ __bf16 sG11t[128 * 128];  // g11 transposed      (z2|z1 step)
  __shared__ __bf16 sG22 [128 * 128];  // g22n row-major
  __shared__ __bf16 sG22t[128 * 128];  // g22n transposed
  __shared__ __bf16 sZ1b[32 * 128], sZ1q[32 * 128];   // z1, z1^2 (bf16)
  __shared__ __bf16 sZ2b[32 * 128], sZ2q[32 * 128];   // z2, z2^2

  const int tid = threadIdx.x, lane = tid & 31, w = tid >> 5;
  const int chain   = blockIdx.x >> 3;           // 0 = prior, 1 = posterior
  const int rowBase = (blockIdx.x & 7) * 32;
  const int niter   = nInitP[0] + 5;             // burn-in + differentiable steps

  for (int i = tid; i < 128 * 128; i += 256){
    int r = i >> 7, c = i & 127;
    float g  = g11[i];
    float gq = -__expf(g22[i]);                  // g22n = -exp(g22)
    sG11 [i] = (__bf16)g;
    sG22 [i] = (__bf16)gq;
    sG11t[c * 128 + r] = (__bf16)g;
    sG22t[c * 128 + r] = (__bf16)gq;
  }
  const unsigned csalt = chain ? 0xABCD1234u : 0x1234ABCDu;
  for (int i = tid; i < 32 * 128; i += 256){
    int r = i >> 7, c = i & 127;
    unsigned base = csalt ^ ((rowBase + r) << 8) ^ c;
    float z1 = gaussf(hmix(base ^ 0x11111111u));
    float z2 = gaussf(hmix(base ^ 0x22222222u));
    sZ1b[i] = (__bf16)z1;  sZ1q[i] = (__bf16)(z1 * z1);
    sZ2b[i] = (__bf16)z2;  sZ2q[i] = (__bf16)(z2 * z2);
  }

  // hoist lambda values for this thread's fixed (row,col) ownership into regs
  const int colL = w * 16 + (lane & 15);
  const int mof  = (lane & 16) ? 8 : 0;
  float lm1v[16], lv1v[16], lm2v[16], lv2v[16];
  #pragma unroll
  for (int t = 0; t < 2; ++t){
    #pragma unroll
    for (int i = 0; i < 8; ++i){
      int grow = rowBase + t * 16 + mof + i;
      if (chain){
        lm1v[t * 8 + i] = mu1 [grow * 128 + colL];
        lv1v[t * 8 + i] = var1[grow * 128 + colL];
        lm2v[t * 8 + i] = mu2 [grow * 128 + colL];
        lv2v[t * 8 + i] = var2[grow * 128 + colL];
      } else {
        lm1v[t * 8 + i] = 0.0f; lv1v[t * 8 + i] = 0.0f;
        lm2v[t * 8 + i] = 0.0f; lv2v[t * 8 + i] = 0.0f;
      }
    }
  }
  __syncthreads();

  float z1f[16], z2f[16];
  for (int it = 0; it < niter; ++it){
    unsigned s0 = csalt ^ (unsigned)(it * 0x9E3779B9u);
    unsigned sA = hmix(s0 ^ 0x51u) << 15;
    unsigned sB = hmix(s0 ^ 0x52u) << 15;

    // All RNG for both half-steps is data-independent: draw it up front so
    // the transcendental VALU co-executes under the WMMA / ds_load latency.
    float2 n1[8], n2[8];
    #pragma unroll
    for (int i = 0; i < 8; i += 2){
      int g0 = rowBase + mof + i;
      int g1 = rowBase + 16 + mof + i;
      n1[i >> 1]       = gauss2(sA ^ (g0 << 7) ^ colL);
      n1[4 + (i >> 1)] = gauss2(sA ^ 0x40000000u ^ (g1 << 7) ^ colL);
      n2[i >> 1]       = gauss2(sB ^ (g0 << 7) ^ colL);
      n2[4 + (i >> 1)] = gauss2(sB ^ 0x40000000u ^ (g1 << 7) ^ colL);
    }

    // z1 | z2 : B = g^T -> row-major arrays give contiguous loads
    gibbs_half(sZ2b, sZ2q, sZ1b, sZ1q, z1f, sG11, sG22,
               lm1v, lv1v, n1, lane, w);
    __syncthreads();
    // z2 | z1 : B = g -> transposed arrays give contiguous loads
    gibbs_half(sZ1b, sZ1q, sZ2b, sZ2q, z2f, sG11t, sG22t,
               lm2v, lv2v, n2, lane, w);
    __syncthreads();
  }

  // final f32 values carried in registers -> global
  float* o1 = chain ? zq1 : zp1;
  float* o2 = chain ? zq2 : zp2;
  #pragma unroll
  for (int i = 0; i < 8; ++i){
    int r0 = rowBase + mof + i;
    int r1 = rowBase + 16 + mof + i;
    o1[r0 * 128 + colL] = z1f[i];
    o1[r1 * 128 + colL] = z1f[8 + i];
    o2[r0 * 128 + colL] = z2f[i];
    o2[r1 * 128 + colL] = z2f[8 + i];
  }
}

// ---------------------------------------------------------------- reductions
__global__ void prep_g22n(const float* __restrict__ g22, float* __restrict__ g22n){
  int i = blockIdx.x * blockDim.x + threadIdx.x;
  if (i < 128 * 128) g22n[i] = -__expf(g22[i]);
}

// per-sample energy: z1^T g11 z2 + (z1^2)^T g22n (z2^2); one block per sample
__global__ __launch_bounds__(128)
void energy_kernel(const float* __restrict__ z1, const float* __restrict__ z2,
                   const float* __restrict__ g11, const float* __restrict__ g22n,
                   float* __restrict__ partial, float sign)
{
  __shared__ float s2[128], s2q[128], red[128];
  const int b = blockIdx.x, t = threadIdx.x;
  float z2t = z2[b * 128 + t];
  s2[t] = z2t; s2q[t] = z2t * z2t;
  __syncthreads();
  float z1t = z1[b * 128 + t];
  float acc = 0.f, accq = 0.f;
  #pragma unroll 8
  for (int j = 0; j < 128; ++j){
    acc  += g11[t * 128 + j]  * s2[j];
    accq += g22n[t * 128 + j] * s2q[j];
  }
  red[t] = z1t * acc + z1t * z1t * accq;
  __syncthreads();
  for (int s = 64; s > 0; s >>= 1){ if (t < s) red[t] += red[t + s]; __syncthreads(); }
  if (t == 0) partial[b] = sign * red[0];
}

__global__ __launch_bounds__(256)
void kl_kernel(const float* __restrict__ mu, const float* __restrict__ var,
               const float* __restrict__ z, float* __restrict__ partial, int n)
{
  __shared__ float red[256];
  const int t = threadIdx.x;
  const int base = blockIdx.x * 1024;
  float acc = 0.f;
  #pragma unroll
  for (int j = 0; j < 4; ++j){
    int i = base + j * 256 + t;
    if (i < n){ float zv = z[i]; acc += mu[i] * zv + var[i] * zv * zv; }
  }
  red[t] = acc; __syncthreads();
  for (int s = 128; s > 0; s >>= 1){ if (t < s) red[t] += red[t + s]; __syncthreads(); }
  if (t == 0) partial[blockIdx.x] = red[0];
}

__global__ __launch_bounds__(256)
void rec_kernel(const float* __restrict__ xr, const float* __restrict__ x,
                float* __restrict__ partial, int n)
{
  __shared__ float red[256];
  const int t = threadIdx.x;
  const int base = blockIdx.x * 1024;
  float acc = 0.f;
  #pragma unroll
  for (int j = 0; j < 4; ++j){
    int i = base + j * 256 + t;
    if (i < n){ float d = xr[i] - x[i]; acc += d * d; }
  }
  red[t] = acc; __syncthreads();
  for (int s = 128; s > 0; s >>= 1){ if (t < s) red[t] += red[t + s]; __syncthreads(); }
  if (t == 0) partial[blockIdx.x] = red[0];
}

__global__ void final_kernel(const float* __restrict__ partial, int n, float* __restrict__ out){
  if (threadIdx.x == 0 && blockIdx.x == 0){
    float s = 0.f;
    for (int i = 0; i < n; ++i) s += partial[i];  // fixed order -> deterministic
    out[0] = s;
  }
}

// ---------------------------------------------------------------- launch glue
extern "C" void kernel_launch(void* const* d_in, const int* in_sizes, int n_in,
                              void* d_out, int out_size, void* d_ws, size_t ws_size,
                              hipStream_t stream)
{
  // setup_inputs() dict order, params nested in insertion order:
  const float* x1   = (const float*)d_in[0];     // 256x1024
  const float* x2   = (const float*)d_in[1];     // 256x2048
  const float* e1W  = (const float*)d_in[2];     // 1024x512
  const float* e1b  = (const float*)d_in[3];
  const float* e1Wm = (const float*)d_in[4];     // 512x128
  const float* e1bm = (const float*)d_in[5];
  const float* e1Wv = (const float*)d_in[6];
  const float* e1bv = (const float*)d_in[7];
  const float* e2W  = (const float*)d_in[8];     // 2048x512
  const float* e2b  = (const float*)d_in[9];
  const float* e2Wm = (const float*)d_in[10];
  const float* e2bm = (const float*)d_in[11];
  const float* e2Wv = (const float*)d_in[12];
  const float* e2bv = (const float*)d_in[13];
  const float* d1W1 = (const float*)d_in[14];    // 128x512
  const float* d1b1 = (const float*)d_in[15];
  const float* d1W2 = (const float*)d_in[16];    // 512x1024
  const float* d1b2 = (const float*)d_in[17];
  const float* d2W1 = (const float*)d_in[18];    // 128x512
  const float* d2b1 = (const float*)d_in[19];
  const float* d2W2 = (const float*)d_in[20];    // 512x2048
  const float* d2b2 = (const float*)d_in[21];
  const float* g11  = (const float*)d_in[22];    // 128x128
  const float* g22  = (const float*)d_in[23];    // 128x128
  const int*   nini = (const int*)d_in[24];

  float* ws = (float*)d_ws;
  float* h1   = ws;              // 131072
  float* mu1  = h1   + 131072;   // 32768
  float* var1 = mu1  + 32768;    // 32768
  float* h2   = var1 + 32768;    // 131072
  float* mu2  = h2   + 131072;   // 32768
  float* var2 = mu2  + 32768;    // 32768
  float* g22n = var2 + 32768;    // 16384
  float* zp1  = g22n + 16384;    // 32768
  float* zp2  = zp1  + 32768;    // 32768
  float* zq1  = zp2  + 32768;    // 32768
  float* zq2  = zq1  + 32768;    // 32768
  float* hd1  = zq2  + 32768;    // 131072
  float* xr1  = hd1  + 131072;   // 262144
  float* hd2  = xr1  + 262144;   // 131072
  float* xr2  = hd2  + 131072;   // 524288
  float* part = xr2  + 524288;   // 1344 partial sums

  dim3 blk(256);
  // --- encoders
  gemm_bias_act<<<dim3(512/64, 256/64), blk, 0, stream>>>(x1, e1W,  e1b,  h1,   256, 1024, 512, ACT_RELU);
  gemm_bias_act<<<dim3(128/64, 256/64), blk, 0, stream>>>(h1, e1Wm, e1bm, mu1,  256, 512,  128, ACT_NONE);
  gemm_bias_act<<<dim3(128/64, 256/64), blk, 0, stream>>>(h1, e1Wv, e1bv, var1, 256, 512,  128, ACT_NEGEXP);
  gemm_bias_act<<<dim3(512/64, 256/64), blk, 0, stream>>>(x2, e2W,  e2b,  h2,   256, 2048, 512, ACT_RELU);
  gemm_bias_act<<<dim3(128/64, 256/64), blk, 0, stream>>>(h2, e2Wm, e2bm, mu2,  256, 512,  128, ACT_NONE);
  gemm_bias_act<<<dim3(128/64, 256/64), blk, 0, stream>>>(h2, e2Wv, e2bv, var2, 256, 512,  128, ACT_NEGEXP);

  prep_g22n<<<dim3(64), blk, 0, stream>>>(g22, g22n);

  // --- Gibbs chains (prior: blocks 0-7, posterior: blocks 8-15)
  gibbs_kernel<<<dim3(16), blk, 0, stream>>>(g11, g22, mu1, var1, mu2, var2,
                                             zp1, zp2, zq1, zq2, nini);

  // --- decoders
  gemm_bias_act<<<dim3(512/64,  256/64), blk, 0, stream>>>(zq1, d1W1, d1b1, hd1, 256, 128, 512,  ACT_RELU);
  gemm_bias_act<<<dim3(1024/64, 256/64), blk, 0, stream>>>(hd1, d1W2, d1b2, xr1, 256, 512, 1024, ACT_NONE);
  gemm_bias_act<<<dim3(512/64,  256/64), blk, 0, stream>>>(zq2, d2W1, d2b1, hd2, 256, 128, 512,  ACT_RELU);
  gemm_bias_act<<<dim3(2048/64, 256/64), blk, 0, stream>>>(hd2, d2W2, d2b2, xr2, 256, 512, 2048, ACT_NONE);

  // --- loss reductions (deterministic two-stage)
  energy_kernel<<<dim3(256), dim3(128), 0, stream>>>(zq1, zq2, g11, g22n, part + 0,   +1.0f);
  energy_kernel<<<dim3(256), dim3(128), 0, stream>>>(zp1, zp2, g11, g22n, part + 256, -1.0f);
  kl_kernel <<<dim3(32),  blk, 0, stream>>>(mu1, var1, zq1, part + 512, 32768);
  kl_kernel <<<dim3(32),  blk, 0, stream>>>(mu2, var2, zq2, part + 544, 32768);
  rec_kernel<<<dim3(256), blk, 0, stream>>>(xr1, x1, part + 576, 262144);
  rec_kernel<<<dim3(512), blk, 0, stream>>>(xr2, x2, part + 832, 524288);
  final_kernel<<<dim3(1), dim3(32), 0, stream>>>(part, 1344, (float*)d_out);
}
